// Embed_36060545417380
// MI455X (gfx1250) — compile-verified, matrix operand order checked by reading, not run on hardware
//
#include <hip/hip_runtime.h>

// CDNA5 / gfx1250 — wave32, WMMA f16->f32
typedef __attribute__((ext_vector_type(16))) _Float16 v16h;
typedef __attribute__((ext_vector_type(8)))  float    v8f;

#define DIMK 64                 // DIM (K of the GEMM)
#define NCOL 64                 // N
#define MTOT 8193               // M = 2*64*64 + 1
#define ROWS 68                 // LDS sort-slab row stride in dwords (64 + 4 pad)

__global__ __launch_bounds__(128)
void embed_sortdot_kernel(const float* __restrict__ mat,   // (64, 64)  d-major
                          const float* __restrict__ ws,    // (M, 64)   m-major
                          const float* __restrict__ wd,    // (M, 64)   m-major
                          float* __restrict__ out)         // (M,)
{
    // mat staged in LDS as f16, PRE-SWIZZLED into WMMA B-fragment order:
    //   sBfrag[((kk*4 + nt)*32 + lane)*16 + j]  (8 KB total)
    // so a lane fetches one whole v16h fragment with 2x ds_load_b128.
    __shared__ __align__(16) _Float16 sBfrag[8 * 32 * 16];
    __shared__ __align__(16) float    sV[4][32 * ROWS];    // 34 KB sort staging

    const int tid  = threadIdx.x;
    const int wave = tid >> 5;
    const int lane = tid & 31;
    const int h    = lane >> 4;        // lane half (0/1)
    const int l15  = lane & 15;

    const int mBlock = blockIdx.x * 128;

    // Prefetch this lane's weights_dot row (global_prefetch_b8) — consumed after the sort.
    {
        int mg = mBlock + wave * 32 + lane;
        int mc = mg < MTOT ? mg : (MTOT - 1);
        __builtin_prefetch(wd + (size_t)mc * NCOL, 0, 0);
    }

    // ---- Stage mat -> LDS f16 in B-fragment order.
    // Coalesced global reads; LDS scatter via the inverse 16-bit-B-operand map.
    #pragma unroll
    for (int r = 0; r < 32; ++r) {
        int e = tid + r * 128;              // linear index into mat (k*64 + n)
        int k = e >> 6;
        int n = e & 63;
        int kk   = k >> 5;
        int g    = (k >> 4) & 1;
        int hh   = (k >> 3) & 1;
        int rem  = k & 7;                   // q = g*4 + rem>>1, pos = rem&1
        int j    = g * 8 + rem;             // half index inside the v16h fragment
        int nt   = n >> 4;
        int ln   = hh * 16 + (n & 15);      // owning lane of this element
        int dest = ((kk * 4 + nt) * 32 + ln) * 16 + j;
        sBfrag[dest] = (_Float16)mat[e];
    }
    __syncthreads();

    // ---- Load all 8 B fragments once per wave (m-invariant), 2x ds_load_b128 each.
    v16h b[2][4];
    #pragma unroll
    for (int kk = 0; kk < 2; ++kk)
        #pragma unroll
        for (int nt = 0; nt < 4; ++nt)
            b[kk][nt] = *(const v16h*)&sBfrag[((kk * 4 + nt) * 32 + lane) * 16];

    // ---- GEMM: each wave computes sort_matmul for 32 m-rows via WMMA ----
    #pragma unroll
    for (int mtile = 0; mtile < 2; ++mtile) {
        int mg = mBlock + wave * 32 + mtile * 16 + l15;
        int mc = mg < MTOT ? mg : (MTOT - 1);
        const float* arow = ws + (size_t)mc * DIMK;

        // A fragments: lane's 8 K-values per (kk,g) are contiguous -> float4 loads.
        v16h a[2];
        #pragma unroll
        for (int kk = 0; kk < 2; ++kk) {
            #pragma unroll
            for (int g = 0; g < 2; ++g) {
                const float4* p = (const float4*)(arow + kk * 32 + g * 16 + h * 8);
                float4 w0 = p[0];
                float4 w1 = p[1];
                a[kk][8 * g + 0] = (_Float16)w0.x;
                a[kk][8 * g + 1] = (_Float16)w0.y;
                a[kk][8 * g + 2] = (_Float16)w0.z;
                a[kk][8 * g + 3] = (_Float16)w0.w;
                a[kk][8 * g + 4] = (_Float16)w1.x;
                a[kk][8 * g + 5] = (_Float16)w1.y;
                a[kk][8 * g + 6] = (_Float16)w1.z;
                a[kk][8 * g + 7] = (_Float16)w1.w;
            }
        }

        #pragma unroll
        for (int nt = 0; nt < 4; ++nt) {
            v8f acc = {};
            acc = __builtin_amdgcn_wmma_f32_16x16x32_f16(
                      false, a[0], false, b[0][nt], (short)0, acc, false, false);
            acc = __builtin_amdgcn_wmma_f32_16x16x32_f16(
                      false, a[1], false, b[1][nt], (short)0, acc, false, false);
            // Scatter D tile to padded LDS slab: VGPR r -> M = r + h*8, N = l15
            #pragma unroll
            for (int r = 0; r < 8; ++r) {
                int ml = mtile * 16 + h * 8 + r;
                int n  = nt * 16 + l15;
                sV[wave][ml * ROWS + n] = acc[r];
            }
        }
    }
    __syncthreads();

    // ---- Per-lane: gather own m-row, bitonic sort (64), dot with weights_dot ----
    float v[64];
    {
        const float* row = &sV[wave][lane * ROWS];
        #pragma unroll
        for (int i = 0; i < 16; ++i) {
            float4 t = *(const float4*)(row + 4 * i);
            v[4 * i]     = t.x;
            v[4 * i + 1] = t.y;
            v[4 * i + 2] = t.z;
            v[4 * i + 3] = t.w;
        }
    }

    // Fully unrolled ascending bitonic sorting network on 64 registers.
    #pragma unroll
    for (int k = 2; k <= 64; k <<= 1) {
        #pragma unroll
        for (int j = k >> 1; j > 0; j >>= 1) {
            #pragma unroll
            for (int i = 0; i < 64; ++i) {
                int ixj = i ^ j;
                if (ixj > i) {
                    bool up = ((i & k) == 0);
                    float x = v[i], y = v[ixj];
                    float lo = fminf(x, y), hi = fmaxf(x, y);
                    v[i]   = up ? lo : hi;
                    v[ixj] = up ? hi : lo;
                }
            }
        }
    }

    int mg = mBlock + wave * 32 + lane;
    int mc = mg < MTOT ? mg : (MTOT - 1);
    const float* wrow = wd + (size_t)mc * NCOL;
    float e = 0.0f;
    #pragma unroll
    for (int i = 0; i < 16; ++i) {
        float4 w = *(const float4*)(wrow + 4 * i);
        e = fmaf(w.x, v[4 * i],     e);
        e = fmaf(w.y, v[4 * i + 1], e);
        e = fmaf(w.z, v[4 * i + 2], e);
        e = fmaf(w.w, v[4 * i + 3], e);
    }
    if (mg < MTOT) out[mg] = e;
}

extern "C" void kernel_launch(void* const* d_in, const int* in_sizes, int n_in,
                              void* d_out, int out_size, void* d_ws, size_t ws_size,
                              hipStream_t stream) {
    const float* mat = (const float*)d_in[0];   // (64, 64) fp32
    const float* ws  = (const float*)d_in[1];   // (8193, 64) fp32
    const float* wd  = (const float*)d_in[2];   // (8193, 64) fp32
    float* out = (float*)d_out;                 // (8193,) fp32

    const int blocks = (MTOT + 127) / 128;      // 65 blocks x 128 threads (4 waves)
    embed_sortdot_kernel<<<blocks, 128, 0, stream>>>(mat, ws, wd, out);
}
